// SelectiveSSM_82755429859939
// MI455X (gfx1250) — matrix operand order repaired and checked by reading, not compiled, not run on hardware
//
#include <hip/hip_runtime.h>
#include <hip/hip_bf16.h>

#define BATCH  2
#define SEQ    2048
#define DMODEL 1024
#define DINNER 2048
#define DSTATE 64
#define DTRANK 64
#define NSSM   192                 // DTRANK + 2*DSTATE
#define MROWS  (BATCH*SEQ)         // 4096

#define BM 256                     // block M tile (8 waves * 32 rows)
#define BN 64                      // block N tile (staged in LDS via TDM)

typedef __attribute__((ext_vector_type(16))) __bf16 v16bf;
typedef __attribute__((ext_vector_type(8)))  float  v8f;
typedef unsigned int v4u __attribute__((ext_vector_type(4)));
typedef int v8i __attribute__((ext_vector_type(8)));
typedef int v4i __attribute__((ext_vector_type(4)));

#define HAVE_TDM __has_builtin(__builtin_amdgcn_tensor_load_to_lds)

struct Bytes32 { uint4 a, b; };

// A fragment (16x32 bf16, row-major source): lane half h holds K = {h*8..h*8+7, h*8+16..h*8+23}
__device__ __forceinline__ v16bf load_frag_a(const __bf16* p) {
  Bytes32 t;
  t.a = *(const uint4*)(p);
  t.b = *(const uint4*)(p + 16);
  return __builtin_bit_cast(v16bf, t);
}
// B fragment (32x16 bf16) from a [16-rows-of-N x 32-K] row-major tile: lane half h holds
// K = {h*8..h*8+7, h*8+16..h*8+23}
__device__ __forceinline__ v16bf load_frag_b(const __bf16* p) {
  Bytes32 t;
  t.a = *(const uint4*)(p);
  t.b = *(const uint4*)(p + 16);
  return __builtin_bit_cast(v16bf, t);
}

__global__ void cvt_f32_bf16_kernel(const float* __restrict__ s,
                                    __bf16* __restrict__ d, int n) {
  int i = blockIdx.x * blockDim.x + threadIdx.x;
  if (i < n) d[i] = (__bf16)s[i];
}

// C[M,N] = A[M,K](bf16) * W[N,K](bf16)^T with f32 accumulation.
// Block = 8 waves: 256x64 tile. B tile (64xN rows, 32 K) staged in LDS by the
// Tensor Data Mover (double-buffered, s_wait_tensorcnt pipelined); A streamed
// from global with b128 loads. Each wave computes 32x64 (2x4 WMMA fragments).
__global__ void gemm_wmma_bf16_kernel(const __bf16* __restrict__ A, int lda,
                                      const __bf16* __restrict__ W, int ldw,
                                      float* __restrict__ C, int ldc,
                                      int M, int N, int K) {
  __shared__ alignas(16) __bf16 bsh[2][BN * 32];

  int wave = threadIdx.x >> 5;
  int lane = threadIdx.x & 31;
  int nb = N / BN;
  int bm = blockIdx.x / nb;
  int bn = blockIdx.x - bm * nb;
  int m0 = bm * BM + wave * 32;
  int n0 = bn * BN;
  int half = lane >> 4;
  int lan  = lane & 15;

  const __bf16* Abase0 = A + (size_t)(m0 + lan) * lda + half * 8;
  const __bf16* Abase1 = Abase0 + (size_t)16 * lda;

  v8f acc[2][4] = {};

#if HAVE_TDM
  // TDM descriptor issue: 2D tensor tile [BN rows x 32 K] of bf16 from W into LDS.
  auto issue = [&](int k0, int buf) {
    unsigned long long ga =
        (unsigned long long)(uintptr_t)(W + (size_t)n0 * ldw + k0);
    unsigned lds_off = (unsigned)(uintptr_t)(&bsh[buf][0]);
    unsigned Krem = (unsigned)(K - k0);
    unsigned Nrem = (unsigned)(N - n0);
    v4u g0;
    g0[0] = 1u;                                        // count=1 valid descriptor
    g0[1] = lds_off;                                   // lds_addr
    g0[2] = (unsigned)ga;                              // global_addr[31:0]
    g0[3] = ((unsigned)(ga >> 32) & 0x01FFFFFFu)       // global_addr[56:32]
            | 0x80000000u;                             // type=2 ("image")
    v8i g1;
    g1[0] = 0x00010000;                                // data_size = 2 bytes
    g1[1] = (int)((Krem & 0xFFFFu) << 16);             // tensor_dim0 lo16
    g1[2] = (int)((Krem >> 16) | ((Nrem & 0xFFFFu) << 16)); // dim0 hi | dim1 lo
    g1[3] = (int)((Nrem >> 16) | (32u << 16));         // dim1 hi | tile_dim0=32
    g1[4] = BN;                                        // tile_dim1=64, tile_dim2=0
    g1[5] = ldw;                                       // tensor_dim0_stride lo32
    g1[6] = 0;                                         // stride hi16 | dim1_stride lo
    g1[7] = 0;
    v4i gz = {0, 0, 0, 0};
#if defined(__clang_major__) && __clang_major__ >= 23
    v8i gz8 = {0, 0, 0, 0, 0, 0, 0, 0};
    __builtin_amdgcn_tensor_load_to_lds(g0, g1, gz, gz, gz8, 0);
#else
    __builtin_amdgcn_tensor_load_to_lds(g0, g1, gz, gz, 0);
#endif
  };
#else
  // Fallback: cooperative copy of the 64x32 bf16 tile (4KB = 256 lanes x 16B).
  auto copy_tile = [&](int k0, int buf) {
    int t   = threadIdx.x;
    int row = t >> 2;          // 0..63
    int c4  = t & 3;           // 16B chunk within 64B row
    const uint4* src =
        (const uint4*)(W + (size_t)(n0 + row) * ldw + k0 + c4 * 8);
    *(uint4*)(&bsh[buf][row * 32 + c4 * 8]) = *src;
  };
#endif

  int nk = K >> 5;
#if HAVE_TDM
  if (wave == 0) issue(0, 0);
#else
  copy_tile(0, 0);
#endif

  for (int i = 0; i < nk; i++) {
    int cur = i & 1;
#if HAVE_TDM
    if (wave == 0) {
      if (i + 1 < nk) {
        issue((i + 1) << 5, cur ^ 1);
        __builtin_amdgcn_s_wait_tensorcnt(1);
      } else {
        __builtin_amdgcn_s_wait_tensorcnt(0);
      }
    }
#endif
    __syncthreads();
#if !HAVE_TDM
    if (i + 1 < nk) copy_tile((i + 1) << 5, cur ^ 1);
#endif
    int k0 = i << 5;
    v16bf a0 = load_frag_a(Abase0 + k0);
    v16bf a1 = load_frag_a(Abase1 + k0);
    #pragma unroll
    for (int j = 0; j < 4; j++) {
      v16bf bj = load_frag_b(&bsh[cur][(j * 16 + lan) * 32 + half * 8]);
      acc[0][j] = __builtin_amdgcn_wmma_f32_16x16x32_bf16(false, a0, false, bj,
                                                          (short)0, acc[0][j], false, false);
      acc[1][j] = __builtin_amdgcn_wmma_f32_16x16x32_bf16(false, a1, false, bj,
                                                          (short)0, acc[1][j], false, false);
    }
    __syncthreads();
  }

  // D layout: vgpr r, lane half h -> m = r + 8*h ; n = lane&15
  #pragma unroll
  for (int i = 0; i < 2; i++)
    #pragma unroll
    for (int j = 0; j < 4; j++) {
      int mrow = m0 + i * 16 + half * 8;
      int ncol = n0 + j * 16 + lan;
      float* Cp = C + (size_t)mrow * ldc + ncol;
      #pragma unroll
      for (int r = 0; r < 8; r++) Cp[(size_t)r * ldc] = acc[i][j][r];
    }
}

__device__ __forceinline__ float silu(float v) {
  return v / (1.f + __expf(-v));
}

// causal depthwise conv (d_conv=4) + bias + SiLU, reading x_s half of xz
__global__ void conv_silu_kernel(const float* __restrict__ xz,
                                 const float* __restrict__ conv_w,
                                 const float* __restrict__ conv_b,
                                 float* __restrict__ xc,
                                 __bf16* __restrict__ xc_bf) {
  int idx = blockIdx.x * blockDim.x + threadIdx.x;
  if (idx >= MROWS * DINNER) return;
  int d = idx & (DINNER - 1);
  int l = (idx >> 11) & (SEQ - 1);
  int b = idx >> 22;
  float acc = conv_b[d];
  #pragma unroll
  for (int j = 0; j < 4; j++) {
    int ls = l - 3 + j;
    if (ls >= 0)
      acc += conv_w[d * 4 + j] * xz[((size_t)(b * SEQ + ls)) * (2 * DINNER) + d];
  }
  float s = silu(acc);
  xc[idx] = s;
  xc_bf[idx] = (__bf16)s;
}

// dt = softplus(dt_pre + b_dt[d]) (in place)
__global__ void softplus_bias_kernel(float* __restrict__ dt,
                                     const float* __restrict__ b_dt) {
  int idx = blockIdx.x * blockDim.x + threadIdx.x;
  if (idx >= MROWS * DINNER) return;
  int d = idx & (DINNER - 1);
  float v = dt[idx] + b_dt[d];
  dt[idx] = (v > 20.f) ? v : log1pf(__expf(v));
}

// Selective scan: one wave per (b, d); lane t keeps h[n=t], h[n=t+32] in registers.
__global__ void scan_kernel(const float* __restrict__ dtv,   // [B*L, DINNER]
                            const float* __restrict__ ssm,   // [B*L, NSSM] (dt|B|C)
                            const float* __restrict__ xc,    // [B*L, DINNER]
                            const float* __restrict__ A_log, // [DINNER, DSTATE]
                            float* __restrict__ y) {         // [B*L, DINNER]
  int gw   = (int)((blockIdx.x * blockDim.x + threadIdx.x) >> 5);
  int lane = threadIdx.x & 31;
  int b = gw >> 11;            // DINNER = 2048
  int d = gw & (DINNER - 1);

  float A0 = -__expf(A_log[(size_t)d * DSTATE + lane]);
  float A1 = -__expf(A_log[(size_t)d * DSTATE + lane + 32]);
  float h0 = 0.f, h1 = 0.f;

  const float* ssm_b = ssm + (size_t)b * SEQ * NSSM;
  const float* dt_b  = dtv + (size_t)b * SEQ * DINNER + d;
  const float* xc_b  = xc  + (size_t)b * SEQ * DINNER + d;
  float*       y_b   = y   + (size_t)b * SEQ * DINNER + d;

  for (int l = 0; l < SEQ; ++l) {
    const float* row = ssm_b + (size_t)l * NSSM;
    __builtin_prefetch(row + NSSM, 0, 1);   // global_prefetch next timestep
    float Bt0 = row[DTRANK + lane];
    float Bt1 = row[DTRANK + lane + 32];
    float Ct0 = row[DTRANK + DSTATE + lane];
    float Ct1 = row[DTRANK + DSTATE + lane + 32];
    float dt_t = dt_b[(size_t)l * DINNER];
    float x_t  = xc_b[(size_t)l * DINNER];
    float dx = dt_t * x_t;
    h0 = __expf(dt_t * A0) * h0 + dx * Bt0;
    h1 = __expf(dt_t * A1) * h1 + dx * Bt1;
    float p = h0 * Ct0 + h1 * Ct1;
    #pragma unroll
    for (int off = 16; off > 0; off >>= 1) p += __shfl_xor(p, off, 32);
    if (lane == 0) y_b[(size_t)l * DINNER] = p;
  }
}

// y = (y + xc*D) * silu(z)  -> bf16 for final GEMM
__global__ void gate_kernel(const float* __restrict__ y,
                            const float* __restrict__ xc,
                            const float* __restrict__ xz,
                            const float* __restrict__ Dp,
                            __bf16* __restrict__ y_bf) {
  int idx = blockIdx.x * blockDim.x + threadIdx.x;
  if (idx >= MROWS * DINNER) return;
  int d   = idx & (DINNER - 1);
  int row = idx >> 11;
  float z = xz[(size_t)row * (2 * DINNER) + DINNER + d];
  float v = (y[idx] + xc[idx] * Dp[d]) * silu(z);
  y_bf[idx] = (__bf16)v;
}

extern "C" void kernel_launch(void* const* d_in, const int* in_sizes, int n_in,
                              void* d_out, int out_size, void* d_ws, size_t ws_size,
                              hipStream_t stream) {
  const float* x      = (const float*)d_in[0];
  const float* W_in   = (const float*)d_in[1];
  const float* conv_w = (const float*)d_in[2];
  const float* conv_b = (const float*)d_in[3];
  const float* W_x    = (const float*)d_in[4];
  const float* W_dt   = (const float*)d_in[5];
  const float* b_dt   = (const float*)d_in[6];
  const float* A_log  = (const float*)d_in[7];
  const float* Dvec   = (const float*)d_in[8];
  const float* W_out  = (const float*)d_in[9];
  float* out = (float*)d_out;

  char* base = (char*)d_ws;
  size_t off = 0;
  auto alloc = [&](size_t bytes) -> void* {
    void* p = base + off;
    off = (off + bytes + 255) & ~(size_t)255;
    return p;
  };

  __bf16* x_bf    = (__bf16*)alloc(sizeof(__bf16) * (size_t)MROWS * DMODEL);
  __bf16* Win_bf  = (__bf16*)alloc(sizeof(__bf16) * (size_t)(2 * DINNER) * DMODEL);
  __bf16* Wx_bf   = (__bf16*)alloc(sizeof(__bf16) * (size_t)NSSM * DINNER);
  __bf16* Wdt_bf  = (__bf16*)alloc(sizeof(__bf16) * (size_t)DINNER * DTRANK);
  __bf16* Wout_bf = (__bf16*)alloc(sizeof(__bf16) * (size_t)DMODEL * DINNER);
  float*  xz      = (float*) alloc(sizeof(float)  * (size_t)MROWS * 2 * DINNER);
  float*  xc      = (float*) alloc(sizeof(float)  * (size_t)MROWS * DINNER);
  __bf16* xc_bf   = (__bf16*)alloc(sizeof(__bf16) * (size_t)MROWS * DINNER);
  float*  ssm     = (float*) alloc(sizeof(float)  * (size_t)MROWS * NSSM);
  __bf16* ssm_bf  = (__bf16*)alloc(sizeof(__bf16) * (size_t)MROWS * NSSM);
  float*  dt      = (float*) alloc(sizeof(float)  * (size_t)MROWS * DINNER);
  float*  yb      = (float*) alloc(sizeof(float)  * (size_t)MROWS * DINNER);
  __bf16* y_bf    = (__bf16*)alloc(sizeof(__bf16) * (size_t)MROWS * DINNER);

  const int T = 256;
  auto cdiv = [](long long a, long long b) { return (int)((a + b - 1) / b); };

  // ---- bf16 conversions of inputs/weights ----
  {
    int n;
    n = MROWS * DMODEL;
    cvt_f32_bf16_kernel<<<cdiv(n, T), T, 0, stream>>>(x, x_bf, n);
    n = 2 * DINNER * DMODEL;
    cvt_f32_bf16_kernel<<<cdiv(n, T), T, 0, stream>>>(W_in, Win_bf, n);
    n = NSSM * DINNER;
    cvt_f32_bf16_kernel<<<cdiv(n, T), T, 0, stream>>>(W_x, Wx_bf, n);
    n = DINNER * DTRANK;
    cvt_f32_bf16_kernel<<<cdiv(n, T), T, 0, stream>>>(W_dt, Wdt_bf, n);
    n = DMODEL * DINNER;
    cvt_f32_bf16_kernel<<<cdiv(n, T), T, 0, stream>>>(W_out, Wout_bf, n);
  }

  auto gemm_blocks = [&](int M, int N) { return (M / BM) * (N / BN); };

  // ---- GEMM1: xz = x @ W_in^T   [4096 x 4096], K=1024 ----
  gemm_wmma_bf16_kernel<<<gemm_blocks(MROWS, 2 * DINNER), T, 0, stream>>>(
      x_bf, DMODEL, Win_bf, DMODEL, xz, 2 * DINNER, MROWS, 2 * DINNER, DMODEL);

  // ---- causal depthwise conv + SiLU ----
  {
    int n = MROWS * DINNER;
    conv_silu_kernel<<<cdiv(n, T), T, 0, stream>>>(xz, conv_w, conv_b, xc, xc_bf);
  }

  // ---- GEMM2: ssm = xc @ W_x^T  [4096 x 192], K=2048 ----
  gemm_wmma_bf16_kernel<<<gemm_blocks(MROWS, NSSM), T, 0, stream>>>(
      xc_bf, DINNER, Wx_bf, DINNER, ssm, NSSM, MROWS, NSSM, DINNER);

  // bf16 copy of ssm (dt_rank part feeds GEMM3)
  {
    int n = MROWS * NSSM;
    cvt_f32_bf16_kernel<<<cdiv(n, T), T, 0, stream>>>(ssm, ssm_bf, n);
  }

  // ---- GEMM3: dt_pre = ssm[:, :64] @ W_dt^T  [4096 x 2048], K=64 ----
  gemm_wmma_bf16_kernel<<<gemm_blocks(MROWS, DINNER), T, 0, stream>>>(
      ssm_bf, NSSM, Wdt_bf, DTRANK, dt, DINNER, MROWS, DINNER, DTRANK);

  // ---- softplus(dt_pre + b_dt) ----
  {
    int n = MROWS * DINNER;
    softplus_bias_kernel<<<cdiv(n, T), T, 0, stream>>>(dt, b_dt);
  }

  // ---- selective scan: 4096 waves (one per (b,d)), 8 waves/block ----
  scan_kernel<<<(BATCH * DINNER) / 8, T, 0, stream>>>(dt, ssm, xc, A_log, yb);

  // ---- skip + gate -> bf16 ----
  {
    int n = MROWS * DINNER;
    gate_kernel<<<cdiv(n, T), T, 0, stream>>>(yb, xc, xz, Dvec, y_bf);
  }

  // ---- GEMM4: out = y @ W_out^T  [4096 x 1024], K=2048 ----
  gemm_wmma_bf16_kernel<<<gemm_blocks(MROWS, DMODEL), T, 0, stream>>>(
      y_bf, DINNER, Wout_bf, DINNER, out, DMODEL, MROWS, DMODEL, DINNER);

  (void)in_sizes; (void)n_in; (void)out_size; (void)ws_size;
}
